// WLHNEncoder_51153060495382
// MI455X (gfx1250) — matrix operand order repaired and checked by reading, not compiled, not run on hardware
//
#include <hip/hip_runtime.h>

// ---------------- problem constants (match reference) ----------------
#define NN      20000      // N_NODES
#define NE      320000     // N_EDGES
#define IN_DIM  256
#define HID     512
#define NL      3          // N_LAYERS
#define NG      64         // N_GRAPHS
#define MINN    1e-15f
#define BNEPS   1e-5f
#define SCALING 0.46211715726000974f    // tanh(TAU/2) = tanh(0.5)
#define PCOMP  (-0.044194173824159216f) // -1/sqrt(512): unit p_vec component

typedef __attribute__((ext_vector_type(16))) __bf16 v16bf;
typedef __attribute__((ext_vector_type(8)))  __bf16 v8bf;
typedef __attribute__((ext_vector_type(4)))  __bf16 v4bf;
typedef __attribute__((ext_vector_type(8)))  float  v8f;
typedef __bf16 bf16_t;

__device__ __forceinline__ float wave_sum(float v) {
#pragma unroll
  for (int m = 16; m >= 1; m >>= 1) v += __shfl_xor(v, m, 32);
  return v;
}

// ---------------- utility: zero a float buffer ----------------
__global__ void zero_f32(float* __restrict__ p, int n) {
  int i = blockIdx.x * blockDim.x + threadIdx.x;
  if (i < n) p[i] = 0.0f;
}

// ---------------- WMMA GEMM: C = relu( f(A) @ W + bias ) ----------------
// MODE 0: f(A)=A; MODE 1: f(A)=A+A2; MODE 2: f(A)=batchnorm-affine(A).
// A: MxK row-major f32, W: KxN row-major f32, C: MxN f32.
// Requires M%32==0, N%128==0, K%64==0 (true for all shapes here).
// Block = 256 threads = 8 waves (2 x 4); block tile 32(M) x 128(N).
// Each wave owns a 16(M) x 32(N) strip: one A fragment, two B fragments,
// 2 WMMAs per 32-wide K-step. LDS double-buffered; fragments are contiguous
// b128 runs:
//   A fragment (lane) = row mloc, K run  [hi*16..+15]
//   B fragment (lane) = col nloc, K run  [hi*16..+15]  (B transposed: Bs[n][k])
template <int MODE>
__global__ __launch_bounds__(256)
void gemm_bias_relu(const float* __restrict__ A, const float* __restrict__ A2,
                    const float* __restrict__ bn_mean, const float* __restrict__ bn_var,
                    const float* __restrict__ bn_gamma, const float* __restrict__ bn_beta,
                    const float* __restrict__ W, const float* __restrict__ bias,
                    float* __restrict__ C, int M, int N, int K) {
  __shared__ __align__(16) bf16_t As[2][32 * 40];   // [m][k], stride 40 halves
  __shared__ __align__(16) bf16_t Bs[2][128 * 40];  // [n][k], stride 40 halves

  const int tid    = threadIdx.x;
  const int lane   = tid & 31;
  const int wave   = tid >> 5;
  const int wm     = wave >> 2;          // 0..1 : 16-row slab within 32
  const int wn     = wave & 3;           // 0..3 : 32-col strip within 128
  const int laneHi = lane >> 4;
  const int laneLo = lane & 15;
  const int bm     = blockIdx.x * 32;
  const int bn     = blockIdx.y * 128;

  // staging indices
  const int ar  = (tid * 4) >> 5;        // A row 0..31
  const int ac  = (tid * 4) & 31;        // A col (K) 0..28 step 4
  const int bnn = tid & 127;             // B col (N) 0..127
  const int bkb = (tid >> 7) * 16;       // B K base 0 or 16

  const int mloc  = wm * 16 + laneLo;
  const int nloc0 = wn * 32 + laneLo;
  const int nloc1 = wn * 32 + 16 + laneLo;

  v8f acc0 = {0.f, 0.f, 0.f, 0.f, 0.f, 0.f, 0.f, 0.f};
  v8f acc1 = {0.f, 0.f, 0.f, 0.f, 0.f, 0.f, 0.f, 0.f};

  auto stage = [&](int buf, int kk) {
    // ---- A tile: 4 contiguous f32 per thread -> one v4bf LDS store ----
    {
      const float4 v = *(const float4*)(A + (size_t)(bm + ar) * K + kk + ac);
      float vv[4] = {v.x, v.y, v.z, v.w};
      if constexpr (MODE == 1) {
        const float4 v2 = *(const float4*)(A2 + (size_t)(bm + ar) * K + kk + ac);
        vv[0] += v2.x; vv[1] += v2.y; vv[2] += v2.z; vv[3] += v2.w;
      }
      if constexpr (MODE == 2) {
#pragma unroll
        for (int j = 0; j < 4; ++j) {
          int f = kk + ac + j;
          vv[j] = (vv[j] - bn_mean[f]) * rsqrtf(bn_var[f] + BNEPS) * bn_gamma[f] + bn_beta[f];
        }
      }
      if (kk + 32 < K)  // prefetch next A K-tile (global_prefetch_b8)
        __builtin_prefetch(A + (size_t)(bm + ar) * K + kk + 32 + ac, 0, 1);
      v4bf pa;
      pa[0] = (bf16_t)vv[0]; pa[1] = (bf16_t)vv[1];
      pa[2] = (bf16_t)vv[2]; pa[3] = (bf16_t)vv[3];
      *(v4bf*)&As[buf][ar * 40 + ac] = pa;
    }
    // ---- B tile transposed: 16 K-strided f32 per thread -> two v8bf stores ----
    {
      v8bf pb0, pb1;
#pragma unroll
      for (int i = 0; i < 8; ++i)
        pb0[i] = (bf16_t)W[(size_t)(kk + bkb + i) * N + bn + bnn];
#pragma unroll
      for (int i = 0; i < 8; ++i)
        pb1[i] = (bf16_t)W[(size_t)(kk + bkb + 8 + i) * N + bn + bnn];
      *(v8bf*)&Bs[buf][bnn * 40 + bkb]     = pb0;
      *(v8bf*)&Bs[buf][bnn * 40 + bkb + 8] = pb1;
    }
  };

  auto compute = [&](int buf) {
    const v8bf a0 = *(const v8bf*)&As[buf][mloc * 40 + laneHi * 16];
    const v8bf a1 = *(const v8bf*)&As[buf][mloc * 40 + laneHi * 16 + 8];
    const v16bf af = __builtin_shufflevector(a0, a1, 0,1,2,3,4,5,6,7,8,9,10,11,12,13,14,15);
    const v8bf b00 = *(const v8bf*)&Bs[buf][nloc0 * 40 + laneHi * 16];
    const v8bf b01 = *(const v8bf*)&Bs[buf][nloc0 * 40 + laneHi * 16 + 8];
    const v8bf b10 = *(const v8bf*)&Bs[buf][nloc1 * 40 + laneHi * 16];
    const v8bf b11 = *(const v8bf*)&Bs[buf][nloc1 * 40 + laneHi * 16 + 8];
    const v16bf bf0 = __builtin_shufflevector(b00, b01, 0,1,2,3,4,5,6,7,8,9,10,11,12,13,14,15);
    const v16bf bf1 = __builtin_shufflevector(b10, b11, 0,1,2,3,4,5,6,7,8,9,10,11,12,13,14,15);
    acc0 = __builtin_amdgcn_wmma_f32_16x16x32_bf16(false, af, false, bf0, (short)0, acc0, false, false);
    acc1 = __builtin_amdgcn_wmma_f32_16x16x32_bf16(false, af, false, bf1, (short)0, acc1, false, false);
  };

  stage(0, 0);
  __syncthreads();
  for (int kk = 0; kk < K; kk += 64) {
    if (kk + 32 < K) stage(1, kk + 32);
    compute(0);
    __syncthreads();
    if (kk + 64 < K) stage(0, kk + 64);
    if (kk + 32 < K) compute(1);
    __syncthreads();
  }

  // ---- epilogue: bias + relu, C/D layout: VGPR r -> M = r + 8*laneHi ----
#pragma unroll
  for (int r = 0; r < 8; ++r) {
    int m = bm + wm * 16 + laneHi * 8 + r;
    int n0 = bn + wn * 32 + laneLo;
    float v0 = acc0[r] + bias[n0];
    float v1 = acc1[r] + bias[n0 + 16];
    C[(size_t)m * N + n0]      = v0 > 0.f ? v0 : 0.f;
    C[(size_t)m * N + n0 + 16] = v1 > 0.f ? v1 : 0.f;
  }
}

// ---------------- edge scatter: agg[dst] += h[src] ----------------
__global__ __launch_bounds__(256)
void scatter_edges(const float* __restrict__ h, const int* __restrict__ src,
                   const int* __restrict__ dst, float* __restrict__ agg) {
  int e = blockIdx.x;
  int t = threadIdx.x;
  int s = src[e], d = dst[e];
  atomicAdd(&agg[(size_t)d * HID + t],        h[(size_t)s * HID + t]);
  atomicAdd(&agg[(size_t)d * HID + t + 256],  h[(size_t)s * HID + t + 256]);
}

// ---------------- batchnorm statistics over nodes ----------------
__global__ __launch_bounds__(256)
void bn_stats(const float* __restrict__ y, float* __restrict__ mean, float* __restrict__ var) {
  __shared__ float ssum[256], ssq[256];
  int f = blockIdx.x;
  float s = 0.f, q = 0.f;
  for (int r = threadIdx.x; r < NN; r += 256) {
    float v = y[(size_t)r * HID + f];
    s += v; q += v * v;
  }
  ssum[threadIdx.x] = s; ssq[threadIdx.x] = q;
  __syncthreads();
  for (int off = 128; off > 0; off >>= 1) {
    if (threadIdx.x < off) {
      ssum[threadIdx.x] += ssum[threadIdx.x + off];
      ssq[threadIdx.x]  += ssq[threadIdx.x + off];
    }
    __syncthreads();
  }
  if (threadIdx.x == 0) {
    float mu = ssum[0] * (1.0f / NN);
    mean[f] = mu;
    var[f]  = ssq[0] * (1.0f / NN) - mu * mu;
  }
}

// ---------------- z init: zA = 0, zB = scaling * h/||h|| ----------------
__global__ __launch_bounds__(256)
void z_init(const float* __restrict__ h, float* __restrict__ zA, float* __restrict__ zB) {
  int node = blockIdx.x * 8 + (threadIdx.x >> 5);
  int lane = threadIdx.x & 31;
  float u[16]; float s = 0.f;
#pragma unroll
  for (int j = 0; j < 16; ++j) {
    u[j] = h[(size_t)node * HID + j * 32 + lane];
    s += u[j] * u[j];
  }
  s = wave_sum(s);
  float inv = SCALING / fmaxf(sqrtf(s), MINN);
#pragma unroll
  for (int j = 0; j < 16; ++j) {
    size_t off = (size_t)node * HID + j * 32 + lane;
    zA[off] = 0.f;
    zB[off] = u[j] * inv;
  }
}

// ---------------- one hyperbolic layer step ----------------
// z_children = scaling*h/||h||; z_parent' = reflect_at_zero(zA, zB);
// z_c' = reflect_through_zero(z_parent', p_vec, z_children);
// zC = reflect_at_zero(z_c', zB).
__global__ __launch_bounds__(256)
void hyper_step(const float* __restrict__ h, const float* __restrict__ zA,
                const float* __restrict__ zB, float* __restrict__ zC) {
  int node = blockIdx.x * 8 + (threadIdx.x >> 5);
  int lane = threadIdx.x & 31;

  float u[16], zc[16], zp[16];
  float su = 0.f, sc = 0.f;
#pragma unroll
  for (int j = 0; j < 16; ++j) {
    size_t off = (size_t)node * HID + j * 32 + lane;
    u[j]  = h[off];
    zc[j] = zB[off];  // z_current
    zp[j] = zA[off];  // z_parent
    su += u[j] * u[j];
    sc += zc[j] * zc[j];
  }
  su = wave_sum(su);
  sc = wave_sum(sc);

  float invu = SCALING / fmaxf(sqrtf(su), MINN);
#pragma unroll
  for (int j = 0; j < 16; ++j) u[j] *= invu;       // u is now z_children

  // reflect_at_zero precompute: a = zc/max(|zc|^2,MINN); r2 = |a|^2 - 1
  float dInv = 1.0f / fmaxf(sc, MINN);
  float a2   = sc * dInv * dInv;
  float r2   = a2 - 1.0f;

  // z_parent' = r2/max(|zp-a|^2) * (zp-a) + a
  float suv = 0.f;
#pragma unroll
  for (int j = 0; j < 16; ++j) {
    zp[j] = zp[j] - zc[j] * dInv;
    suv += zp[j] * zp[j];
  }
  suv = wave_sum(suv);
  float sfac = r2 / fmaxf(suv, MINN);
#pragma unroll
  for (int j = 0; j < 16; ++j) zp[j] = sfac * zp[j] + zc[j] * dInv;  // zp = z_parent'

  // reflect_through_zero(zp, p_vec, u): r = p_vec - zp/||zp||
  float spp = 0.f;
#pragma unroll
  for (int j = 0; j < 16; ++j) spp += zp[j] * zp[j];
  spp = wave_sum(spp);
  float invp = 1.0f / fmaxf(sqrtf(spp), MINN);
  float rx = 0.f, rr = 0.f;
#pragma unroll
  for (int j = 0; j < 16; ++j) {
    float rj = PCOMP - zp[j] * invp;
    rx += rj * u[j];
    rr += rj * rj;
  }
  rx = wave_sum(rx);
  rr = wave_sum(rr);
  float mfac = rx / rr;
#pragma unroll
  for (int j = 0; j < 16; ++j) {
    float rj = PCOMP - zp[j] * invp;
    u[j] = u[j] - 2.0f * rj * mfac;                 // u = z_children reflected
  }

  // z_all = reflect_at_zero(u, zB)  (same a, r2)
  float suv2 = 0.f;
#pragma unroll
  for (int j = 0; j < 16; ++j) {
    float t = u[j] - zc[j] * dInv;
    suv2 += t * t;
  }
  suv2 = wave_sum(suv2);
  float sfac2 = r2 / fmaxf(suv2, MINN);
#pragma unroll
  for (int j = 0; j < 16; ++j) {
    size_t off = (size_t)node * HID + j * 32 + lane;
    zC[off] = sfac2 * (u[j] - zc[j] * dInv) + zc[j] * dInv;
  }
}

// ---------------- logmap0 + segment_sum over graphs ----------------
__global__ __launch_bounds__(256)
void logmap_pool(const float* __restrict__ z, const int* __restrict__ batch,
                 float* __restrict__ out) {
  int node = blockIdx.x * 8 + (threadIdx.x >> 5);
  int lane = threadIdx.x & 31;
  float y[16]; float s = 0.f;
#pragma unroll
  for (int j = 0; j < 16; ++j) {
    y[j] = z[(size_t)node * HID + j * 32 + lane];
    s += y[j] * y[j];
  }
  s = wave_sum(s);
  float yn = fmaxf(sqrtf(s), MINN);
  float c  = fminf(yn, 0.9999999f);
  float t  = 0.5f * logf((1.0f + c) / (1.0f - c));   // atanh
  float scale = t / yn;
  int g = batch[node];
#pragma unroll
  for (int j = 0; j < 16; ++j)
    atomicAdd(&out[(size_t)g * HID + j * 32 + lane], y[j] * scale);
}

// ---------------- driver ----------------
extern "C" void kernel_launch(void* const* d_in, const int* in_sizes, int n_in,
                              void* d_out, int out_size, void* d_ws, size_t ws_size,
                              hipStream_t stream) {
  const float* x     = (const float*)d_in[0];
  const int*   edge  = (const int*)d_in[1];
  const int*   batch = (const int*)d_in[2];
  const float* W0    = (const float*)d_in[3];
  const float* b0    = (const float*)d_in[4];
  const float* W1    = (const float*)d_in[5];
  const float* b1    = (const float*)d_in[6];
  const float* gam   = (const float*)d_in[7];
  const float* bet   = (const float*)d_in[8];
  const float* W2    = (const float*)d_in[9];
  const float* b2    = (const float*)d_in[10];
  float* out = (float*)d_out;

  const size_t NS = (size_t)NN * HID;
  float* h    = (float*)d_ws;
  float* agg  = h + NS;
  float* y    = agg + NS;
  float* zA   = y + NS;
  float* zB   = zA + NS;
  float* zC   = zB + NS;
  float* mean = zC + NS;
  float* var  = mean + HID;

  const int* src = edge;
  const int* dst = edge + NE;

  dim3 gdim(NN / 32, HID / 128);   // (625, 4)
  dim3 bdim(256);

  // h = relu(x @ W0 + b0)
  gemm_bias_relu<0><<<gdim, bdim, 0, stream>>>(x, nullptr, nullptr, nullptr, nullptr, nullptr,
                                               W0, b0, h, NN, HID, IN_DIM);
  // z0 = 0, z1 = scaling * h/||h||
  z_init<<<NN / 8, 256, 0, stream>>>(h, zA, zB);

  for (int i = 0; i < NL; ++i) {
    zero_f32<<<(int)((NS + 255) / 256), 256, 0, stream>>>(agg, (int)NS);
    scatter_edges<<<NE, 256, 0, stream>>>(h, src, dst, agg);
    // y = relu((h+agg) @ W1[i] + b1[i])
    gemm_bias_relu<1><<<gdim, bdim, 0, stream>>>(h, agg, nullptr, nullptr, nullptr, nullptr,
                                                 W1 + (size_t)i * HID * HID, b1 + (size_t)i * HID,
                                                 y, NN, HID, HID);
    bn_stats<<<HID, 256, 0, stream>>>(y, mean, var);
    // h = relu(batchnorm(y) @ W2[i] + b2[i])
    gemm_bias_relu<2><<<gdim, bdim, 0, stream>>>(y, nullptr, mean, var,
                                                 gam + (size_t)i * HID, bet + (size_t)i * HID,
                                                 W2 + (size_t)i * HID * HID, b2 + (size_t)i * HID,
                                                 h, NN, HID, HID);
    hyper_step<<<NN / 8, 256, 0, stream>>>(h, zA, zB, zC);
    float* t = zA; zA = zB; zB = zC; zC = t;
  }

  zero_f32<<<(out_size + 255) / 256, 256, 0, stream>>>(out, out_size);
  logmap_pool<<<NN / 8, 256, 0, stream>>>(zB, batch, out);
}
